// LSTM_GAT_Model_82952998355877
// MI455X (gfx1250) — compile-verified, matrix operand order checked by reading, not run on hardware
//
#include <hip/hip_runtime.h>
#include <hip/hip_bf16.h>
#include <stdint.h>

#define DEV __device__ __forceinline__

typedef __attribute__((ext_vector_type(16))) __bf16 v16bf;
typedef __attribute__((ext_vector_type(8)))  float  v8f;

// ---------------- problem constants (match reference) ----------------
constexpr int N_NODES = 20000;
constexpr int WINDOW  = 60;
constexpr int N_FEAT  = 16;
constexpr int HID     = 128;
constexpr int HEADS   = 4;
constexpr int N_EDGES = 250000;
constexpr int GATES   = 4 * HID;        // 512
constexpr int M_TILE  = 32;             // nodes per block
constexpr int NT      = GATES / 16;     // 32 n-tiles of 16
constexpr int KT_H    = HID / 32;       // 4 k-tiles of 32 (bf16 WMMA K)
constexpr int ET      = N_EDGES + N_NODES; // edges + self loops

// fragment sizes (u32 elements)
constexpr int WHH_FRAG_U32 = NT * KT_H * 256;  // 32768
constexpr int WIH_FRAG_U32 = NT * 256;         // 8192

// ---------------- helpers ----------------
DEV unsigned short f2bf(float f) {
  unsigned u = __float_as_uint(f);
  unsigned r = u + 0x7FFFu + ((u >> 16) & 1u);   // RNE truncate
  return (unsigned short)(r >> 16);
}
DEV float bf2f(unsigned short h) { return __uint_as_float(((unsigned)h) << 16); }
DEV unsigned ordf(float f) {
  unsigned u = __float_as_uint(f);
  return (u & 0x80000000u) ? ~u : (u | 0x80000000u);
}
DEV float unordf(unsigned u) {
  unsigned b = (u & 0x80000000u) ? (u & 0x7FFFFFFFu) : ~u;
  return __uint_as_float(b);
}
DEV float sigmoidf(float x) { return 1.0f / (1.0f + __expf(-x)); }

// CDNA5 async global->LDS copy (16B per lane), tracked by ASYNCcnt
DEV void async_copy16(unsigned lds_off, const void* gaddr) {
  asm volatile("global_load_async_to_lds_b128 %0, %1, off"
               :: "v"(lds_off), "v"(gaddr) : "memory");
}
DEV void wait_async() {
  asm volatile("s_wait_asynccnt 0x0" ::: "memory");
}

union Frag {
  uint4  q[2];
  v16bf  bf;
};

DEV v8f wmma_bf16(const Frag& a, const Frag& b, v8f c) {
  return __builtin_amdgcn_wmma_f32_16x16x32_bf16(false, a.bf, false, b.bf,
                                                 (short)0, c, false, false);
}

// ---------------- prep kernels ----------------
__global__ void kbias(const float* __restrict__ bi, const float* __restrict__ bh,
                      float* __restrict__ bc) {
  int i = blockIdx.x * 256 + threadIdx.x;
  if (i < GATES) bc[i] = bi[i] + bh[i];
}

__global__ void kxbf(const float* __restrict__ x, unsigned short* __restrict__ xbf) {
  size_t i = (size_t)blockIdx.x * 256 + threadIdx.x;
  if (i < (size_t)N_NODES * WINDOW * N_FEAT) xbf[i] = f2bf(x[i]);
}

// pack W [512 x K] into B-fragment layout:
// frag[((ntile*KT + kt)*32 + lane)*8 + v]  u32 = {bf16(k), bf16(k+1)}
__global__ void kfrag(const float* __restrict__ W, unsigned* __restrict__ frag, int K) {
  int KT = K / 32;
  int total = NT * KT * 256;
  int idx = blockIdx.x * 256 + threadIdx.x;
  if (idx >= total) return;
  int v    = idx & 7;
  int lane = (idx >> 3) & 31;
  int kt   = (idx >> 8) % KT;
  int nt   = (idx >> 8) / KT;
  int n     = nt * 16 + (lane & 15);
  int khalf = (lane < 16) ? 0 : 8;
  int kk    = kt * 32 + ((v >= 4) ? 16 : 0) + khalf + 2 * (v & 3);
  unsigned lo = f2bf(W[(size_t)n * K + kk]);
  unsigned hi = f2bf(W[(size_t)n * K + kk + 1]);
  frag[idx] = lo | (hi << 16);
}

// W_ih is [512 x 16]; single K=32 tile, zero-padded above k=15
__global__ void kfrag_ih(const float* __restrict__ W, unsigned* __restrict__ frag) {
  int idx = blockIdx.x * 256 + threadIdx.x;
  if (idx >= WIH_FRAG_U32) return;
  int v    = idx & 7;
  int lane = (idx >> 3) & 31;
  int nt   = idx >> 8;
  int n     = nt * 16 + (lane & 15);
  int khalf = (lane < 16) ? 0 : 8;
  int kk    = ((v >= 4) ? 16 : 0) + khalf + 2 * (v & 3);
  unsigned lo = (kk     < N_FEAT) ? (unsigned)f2bf(W[(size_t)n * N_FEAT + kk])     : 0u;
  unsigned hi = (kk + 1 < N_FEAT) ? (unsigned)f2bf(W[(size_t)n * N_FEAT + kk + 1]) : 0u;
  frag[idx] = lo | (hi << 16);
}

// ---------------- LSTM: WMMA recurrence, cell state fully in registers ----------------
// wave mapping: mtile = wave&1 ; jbase = (wave>>1)&3 ; j in {jbase, jbase+4}
// for each j, the wave owns gate tiles ntile = {j, j+8, j+16, j+24} = (i,f,g,o)
__global__ void klstm(const unsigned short* __restrict__ xbf,
                      const unsigned* __restrict__ whh_then_wih_frag, // contiguous
                      const float* __restrict__ bias,
                      unsigned short* __restrict__ h_out) {
  extern __shared__ char smem[];
  unsigned*       s_whh = (unsigned*)smem;                       // 131072 B
  unsigned*       s_wih = s_whh + WHH_FRAG_U32;                  //  32768 B
  unsigned short* s_h   = (unsigned short*)(s_wih + WIH_FRAG_U32); // 8192 B
  unsigned short* s_x   = s_h + M_TILE * HID;                    //   2048 B

  const int tid   = threadIdx.x;
  const int lane  = tid & 31;
  const int wave  = tid >> 5;
  const int node0 = blockIdx.x * M_TILE;
  const int khA   = (lane < 16) ? 0 : 8;
  const int mtile = wave & 1;
  const int jbase = (wave >> 1) & 3;
  const int mrow  = mtile * 16 + (lane & 15);

  // bulk async copy of both weight-fragment arrays (contiguous, 163840 B)
  {
    const char* src = (const char*)whh_then_wih_frag;
    unsigned    dst = (unsigned)(uintptr_t)s_whh;
    const int nchunk = (WHH_FRAG_U32 + WIH_FRAG_U32) / 4;  // 16B chunks
    for (int c = tid; c < nchunk; c += 256)
      async_copy16(dst + c * 16, src + (size_t)c * 16);
  }
  for (int i = tid; i < M_TILE * HID; i += 256) s_h[i] = 0;
  for (int i = tid; i < M_TILE * 32;  i += 256) s_x[i] = 0;  // K pad stays 0
  wait_async();
  __syncthreads();

  // per-wave persistent registers: biases and cell state for the 2 j-groups
  float bR[2][4];
  #pragma unroll
  for (int q = 0; q < 2; ++q) {
    int jcol = (jbase + 4 * q) * 16 + (lane & 15);
    bR[q][0] = bias[jcol];
    bR[q][1] = bias[128 + jcol];
    bR[q][2] = bias[256 + jcol];
    bR[q][3] = bias[384 + jcol];
  }
  v8f c0 = {}, c1 = {};

  for (int t = 0; t < WINDOW; ++t) {
    // stage x_t tile (already bf16): one async 32B copy per node row by wave 0
    if (wave == 0) {
      unsigned dst = (unsigned)(uintptr_t)(&s_x[lane * 32]);
      const char* src = (const char*)&xbf[((size_t)(node0 + lane) * WINDOW + t) * N_FEAT];
      async_copy16(dst,      src);
      async_copy16(dst + 16, src + 16);
      wait_async();
    }
    __syncthreads();

    // load all A-fragments for this wave's mtile once (kept in VGPRs)
    Frag ax, ah[KT_H];
    {
      const unsigned short* ap = &s_x[mrow * 32 + khA];
      ax.q[0] = *(const uint4*)ap;
      ax.q[1] = *(const uint4*)(ap + 16);
      #pragma unroll
      for (int kt = 0; kt < KT_H; ++kt) {
        const unsigned short* hp = &s_h[mrow * HID + kt * 32 + khA];
        ah[kt].q[0] = *(const uint4*)hp;
        ah[kt].q[1] = *(const uint4*)(hp + 16);
      }
    }
    __syncthreads();   // all A loads done before anyone overwrites s_h below

    #pragma unroll
    for (int q = 0; q < 2; ++q) {
      const int j = jbase + 4 * q;
      v8f acc[4];
      #pragma unroll
      for (int g = 0; g < 4; ++g) {          // gate tiles i,f,g,o
        const int ntile = g * 8 + j;
        Frag b;
        const unsigned* bp = &s_wih[ntile * 256 + lane * 8];
        b.q[0] = *(const uint4*)bp;
        b.q[1] = *(const uint4*)(bp + 4);
        v8f a2 = {};
        a2 = wmma_bf16(ax, b, a2);
        #pragma unroll
        for (int kt = 0; kt < KT_H; ++kt) {
          const unsigned* wp = &s_whh[(ntile * KT_H + kt) * 256 + lane * 8];
          b.q[0] = *(const uint4*)wp;
          b.q[1] = *(const uint4*)(wp + 4);
          a2 = wmma_bf16(ah[kt], b, a2);
        }
        acc[g] = a2;
      }
      // in-register LSTM cell update; h -> LDS as bf16 for next step's A
      v8f c_old = q ? c1 : c0;
      v8f c_new;
      const int jcol = j * 16 + (lane & 15);
      #pragma unroll
      for (int v = 0; v < 8; ++v) {
        float gi = acc[0][v] + bR[q][0];
        float gf = acc[1][v] + bR[q][1];
        float gg = acc[2][v] + bR[q][2];
        float go = acc[3][v] + bR[q][3];
        float c  = sigmoidf(gf) * c_old[v] + sigmoidf(gi) * tanhf(gg);
        c_new[v] = c;
        int mm = mtile * 16 + v + ((lane < 16) ? 0 : 8);
        s_h[mm * HID + jcol] = f2bf(sigmoidf(go) * tanhf(c));
      }
      if (q) c1 = c_new; else c0 = c_new;
    }
  }
  __syncthreads();
  for (int i = tid; i < M_TILE * HID; i += 256)
    h_out[(size_t)node0 * HID + i] = s_h[i];
}

// ---------------- GAT projections: out = h @ W^T + b (bf16 out) ----------------
__global__ void kgemm512(const unsigned short* __restrict__ h,
                         const unsigned* __restrict__ wfrag,
                         const float* __restrict__ bias512,
                         unsigned short* __restrict__ out) {
  extern __shared__ char smem[];
  unsigned*       s_w = (unsigned*)smem;                        // 131072 B
  unsigned short* s_h = (unsigned short*)(s_w + WHH_FRAG_U32);  //   8192 B

  const int tid   = threadIdx.x;
  const int lane  = tid & 31;
  const int wave  = tid >> 5;
  const int node0 = blockIdx.x * M_TILE;
  const int khA   = (lane < 16) ? 0 : 8;
  const int mtile = wave & 1;
  const int nbase = (wave >> 1) * 8;
  const int mrow  = mtile * 16 + (lane & 15);

  // async fill: weight fragments + h tile
  {
    unsigned dstw = (unsigned)(uintptr_t)s_w;
    const char* srcw = (const char*)wfrag;
    for (int c = tid; c < WHH_FRAG_U32 / 4; c += 256)
      async_copy16(dstw + c * 16, srcw + (size_t)c * 16);
    unsigned dsth = (unsigned)(uintptr_t)s_h;
    const char* srch = (const char*)&h[(size_t)node0 * HID];
    for (int c = tid; c < (M_TILE * HID * 2) / 16; c += 256)
      async_copy16(dsth + c * 16, srch + (size_t)c * 16);
  }
  wait_async();
  __syncthreads();

  Frag ah[KT_H];
  #pragma unroll
  for (int kt = 0; kt < KT_H; ++kt) {
    const unsigned short* hp = &s_h[mrow * HID + kt * 32 + khA];
    ah[kt].q[0] = *(const uint4*)hp;
    ah[kt].q[1] = *(const uint4*)(hp + 16);
  }

  for (int i = 0; i < 8; ++i) {
    const int ntile = nbase + i;
    v8f acc = {};
    #pragma unroll
    for (int kt = 0; kt < KT_H; ++kt) {
      Frag b;
      const unsigned* bp = &s_w[(ntile * KT_H + kt) * 256 + lane * 8];
      b.q[0] = *(const uint4*)bp;
      b.q[1] = *(const uint4*)(bp + 4);
      acc = wmma_bf16(ah[kt], b, acc);
    }
    const int n = ntile * 16 + (lane & 15);
    const float bv = bias512[n];
    #pragma unroll
    for (int v = 0; v < 8; ++v) {
      int mm = mtile * 16 + v + ((lane < 16) ? 0 : 8);
      out[((size_t)(node0 + mm)) * GATES + n] = f2bf(acc[v] + bv);
    }
  }
}

// ---------------- edge phase ----------------
__global__ void kscore(const int* __restrict__ ei,
                       const unsigned short* __restrict__ xl,
                       const unsigned short* __restrict__ xr,
                       const float* __restrict__ att,
                       float* __restrict__ scores,
                       unsigned* __restrict__ mbuf) {
  int e = blockIdx.x * 8 + (threadIdx.x >> 5);
  if (e >= ET) return;
  int lane = threadIdx.x & 31;
  int src, dst;
  if (e < N_EDGES) { src = ei[e]; dst = ei[N_EDGES + e]; }
  else             { src = dst = e - N_EDGES; }
  int hh = lane >> 3, cb = lane & 7;
  size_t so  = (size_t)src * GATES + hh * HID;
  size_t dso = (size_t)dst * GATES + hh * HID;
  __builtin_prefetch(&xl[so], 0, 0);
  __builtin_prefetch(&xr[dso], 0, 0);
  float p = 0.0f;
  #pragma unroll 4
  for (int j = 0; j < 16; ++j) {
    int c = cb + j * 8;
    float a = bf2f(xl[so + c]) + bf2f(xr[dso + c]);
    a = (a > 0.0f) ? a : 0.2f * a;              // LeakyReLU(0.2)
    p += att[hh * HID + c] * a;
  }
  p += __shfl_xor(p, 4, 8);
  p += __shfl_xor(p, 2, 8);
  p += __shfl_xor(p, 1, 8);
  if (cb == 0) {
    scores[(size_t)e * HEADS + hh] = p;
    atomicMax(&mbuf[(size_t)dst * HEADS + hh], ordf(p));
  }
}

__global__ void kexp(const int* __restrict__ ei,
                     const float* __restrict__ scores,
                     const unsigned* __restrict__ mbuf,
                     float* __restrict__ exbuf,
                     float* __restrict__ denom) {
  int idx = blockIdx.x * 256 + threadIdx.x;
  if (idx >= ET * HEADS) return;
  int e = idx >> 2, hh = idx & 3;
  int dst = (e < N_EDGES) ? ei[N_EDGES + e] : (e - N_EDGES);
  float m  = unordf(mbuf[(size_t)dst * HEADS + hh]);
  float ex = __expf(scores[idx] - m);
  exbuf[idx] = ex;
  atomicAdd(&denom[(size_t)dst * HEADS + hh], ex);
}

__global__ void kagg(const int* __restrict__ ei,
                     const unsigned short* __restrict__ xl,
                     const float* __restrict__ exbuf,
                     const float* __restrict__ denom,
                     float* __restrict__ outagg) {
  int e = blockIdx.x * 8 + (threadIdx.x >> 5);
  if (e >= ET) return;
  int lane = threadIdx.x & 31;
  int src, dst;
  if (e < N_EDGES) { src = ei[e]; dst = ei[N_EDGES + e]; }
  else             { src = dst = e - N_EDGES; }
  __builtin_prefetch(&xl[(size_t)src * GATES], 0, 0);
  for (int f = lane; f < GATES; f += 32) {
    int hh = f >> 7;
    float alpha = exbuf[(size_t)e * HEADS + hh] / denom[(size_t)dst * HEADS + hh];
    atomicAdd(&outagg[(size_t)dst * GATES + f], alpha * bf2f(xl[(size_t)src * GATES + f]));
  }
}

__global__ void kfinal(const float* __restrict__ outagg,
                       const float* __restrict__ gat_bias,
                       const float* __restrict__ W_pred,
                       const float* __restrict__ b_pred,
                       float* __restrict__ out) {
  int node = blockIdx.x * 8 + (threadIdx.x >> 5);
  if (node >= N_NODES) return;
  int lane = threadIdx.x & 31;
  float acc = 0.0f;
  #pragma unroll
  for (int j = 0; j < 4; ++j) {
    int c = lane + j * 32;
    float v = 0.0f;
    #pragma unroll
    for (int hh = 0; hh < HEADS; ++hh) v += outagg[(size_t)node * GATES + hh * HID + c];
    v = v * 0.25f + gat_bias[c];                 // head mean + bias
    v = (v > 0.0f) ? v : (__expf(v) - 1.0f);     // ELU
    acc += v * W_pred[c];
  }
  acc += __shfl_xor(acc, 16);
  acc += __shfl_xor(acc, 8);
  acc += __shfl_xor(acc, 4);
  acc += __shfl_xor(acc, 2);
  acc += __shfl_xor(acc, 1);
  if (lane == 0) out[node] = acc + b_pred[0];
}

// ---------------- host launcher ----------------
extern "C" void kernel_launch(void* const* d_in, const int* in_sizes, int n_in,
                              void* d_out, int out_size, void* d_ws, size_t ws_size,
                              hipStream_t stream) {
  const float* x      = (const float*)d_in[0];
  const int*   ei     = (const int*)d_in[1];
  const float* W_ih   = (const float*)d_in[2];
  const float* W_hh   = (const float*)d_in[3];
  const float* b_ih   = (const float*)d_in[4];
  const float* b_hh   = (const float*)d_in[5];
  const float* W_l    = (const float*)d_in[6];
  const float* b_l    = (const float*)d_in[7];
  const float* W_r    = (const float*)d_in[8];
  const float* b_r    = (const float*)d_in[9];
  const float* att    = (const float*)d_in[10];
  const float* g_bias = (const float*)d_in[11];
  const float* W_pred = (const float*)d_in[12];
  const float* b_pred = (const float*)d_in[13];

  // workspace carving (256B aligned; all sizes multiple of 256)
  size_t off = 0;
  auto take = [&](size_t bytes) -> char* {
    char* p = (char*)d_ws + off;
    off += (bytes + 255) & ~(size_t)255;
    return p;
  };
  unsigned*       whh_f  = (unsigned*)take(WHH_FRAG_U32 * 4);   // must stay
  unsigned*       wih_f  = (unsigned*)take(WIH_FRAG_U32 * 4);   // contiguous
  unsigned*       wl_f   = (unsigned*)take(WHH_FRAG_U32 * 4);
  unsigned*       wr_f   = (unsigned*)take(WHH_FRAG_U32 * 4);
  float*          bias_c = (float*)take(GATES * 4);
  unsigned short* xbf    = (unsigned short*)take((size_t)N_NODES * WINDOW * N_FEAT * 2);
  unsigned short* h_bf   = (unsigned short*)take((size_t)N_NODES * HID * 2);
  unsigned short* xl     = (unsigned short*)take((size_t)N_NODES * GATES * 2);
  unsigned short* xr     = (unsigned short*)take((size_t)N_NODES * GATES * 2);
  float*          scores = (float*)take((size_t)ET * HEADS * 4);
  float*          exbuf  = (float*)take((size_t)ET * HEADS * 4);
  unsigned*       mbuf   = (unsigned*)take((size_t)N_NODES * HEADS * 4);
  float*          denom  = (float*)take((size_t)N_NODES * HEADS * 4);
  float*          outagg = (float*)take((size_t)N_NODES * GATES * 4);

  constexpr size_t LSTM_LDS = (WHH_FRAG_U32 + WIH_FRAG_U32) * 4   // weights
                            + (size_t)M_TILE * HID * 2            // h (bf16)
                            + (size_t)M_TILE * 32 * 2;            // x tile
  constexpr size_t GEMM_LDS = WHH_FRAG_U32 * 4 + (size_t)M_TILE * HID * 2;

  (void)hipFuncSetAttribute((const void*)klstm,
                            hipFuncAttributeMaxDynamicSharedMemorySize, (int)LSTM_LDS);
  (void)hipFuncSetAttribute((const void*)kgemm512,
                            hipFuncAttributeMaxDynamicSharedMemorySize, (int)GEMM_LDS);

  // prep
  kbias<<<2, 256, 0, stream>>>(b_ih, b_hh, bias_c);
  kxbf<<<(int)(((size_t)N_NODES * WINDOW * N_FEAT + 255) / 256), 256, 0, stream>>>(x, xbf);
  kfrag_ih<<<(WIH_FRAG_U32 + 255) / 256, 256, 0, stream>>>(W_ih, wih_f);
  kfrag<<<(WHH_FRAG_U32 + 255) / 256, 256, 0, stream>>>(W_hh, whh_f, HID);
  kfrag<<<(WHH_FRAG_U32 + 255) / 256, 256, 0, stream>>>(W_l, wl_f, HID);
  kfrag<<<(WHH_FRAG_U32 + 255) / 256, 256, 0, stream>>>(W_r, wr_f, HID);

  // zero segment-max / denominator / aggregation buffers (contiguous)
  size_t zero_bytes = ((char*)outagg + (size_t)N_NODES * GATES * 4) - (char*)mbuf;
  hipMemsetAsync(mbuf, 0, zero_bytes, stream);

  // temporal LSTM (WMMA + async LDS staging)
  klstm<<<N_NODES / M_TILE, 256, LSTM_LDS, stream>>>(xbf, whh_f, bias_c, h_bf);

  // GATv2 linear projections (WMMA)
  kgemm512<<<N_NODES / M_TILE, 256, GEMM_LDS, stream>>>(h_bf, wl_f, b_l, xl);
  kgemm512<<<N_NODES / M_TILE, 256, GEMM_LDS, stream>>>(h_bf, wr_f, b_r, xr);

  // edge attention
  kscore<<<ET / 8, 256, 0, stream>>>(ei, xl, xr, att, scores, mbuf);
  kexp<<<(ET * HEADS + 255) / 256, 256, 0, stream>>>(ei, scores, mbuf, exbuf, denom);
  kagg<<<ET / 8, 256, 0, stream>>>(ei, xl, exbuf, denom, outagg);

  // head mean + bias + ELU + predictor
  kfinal<<<N_NODES / 8, 256, 0, stream>>>(outagg, g_bias, W_pred, b_pred, (float*)d_out);
}
